// multi_gaussian_decoder_58566174049031
// MI455X (gfx1250) — matrix operand order repaired and checked by reading, not compiled, not run on hardware
//
#include <hip/hip_runtime.h>
#include <hip/hip_fp16.h>
#include <hip/hip_bf16.h>

typedef __attribute__((ext_vector_type(16))) _Float16 v16h;
typedef __attribute__((ext_vector_type(8)))  _Float16 v8h;
typedef __attribute__((ext_vector_type(8)))  float    v8f;

typedef unsigned int u32x4 __attribute__((ext_vector_type(4)));
typedef int          i32x4 __attribute__((ext_vector_type(4)));
typedef int          i32x8 __attribute__((ext_vector_type(8)));

#define BOX    256
#define BSZ    16
#define NPTS   20000
#define NCH    2
#define NLAYER 4
#define HID    64
#define LAT    16

#if defined(__has_builtin)
#  if __has_builtin(__builtin_amdgcn_tensor_load_to_lds) && \
      __has_builtin(__builtin_amdgcn_s_wait_tensorcnt)
#    define USE_TDM 1
#  endif
#endif
#ifndef USE_TDM
#  define USE_TDM 0
#endif

// ---------------------------------------------------------------------------
// WMMA fragment helpers (CDNA5 16-bit layouts per ISA 7.12.2, wave32)
// ---------------------------------------------------------------------------

// A-matrix 16x32 f16, source row-major [16][stride64] in LDS.
__device__ __forceinline__ v16h afrag(const _Float16* H, int k0, int lane) {
  int row = lane & 15;
  int off = (lane >> 4) * 8;
  v8h lo = *(const v8h*)(H + row * 64 + k0 + off);
  v8h hi = *(const v8h*)(H + row * 64 + k0 + 16 + off);
  v16h a;
#pragma unroll
  for (int i = 0; i < 8; ++i) { a[i] = lo[i]; a[i + 8] = hi[i]; }
  return a;
}

// B-matrix 32x16 f16 (B = W^T, stored as W row-major [N][kstride]).
__device__ __forceinline__ v16h bfrag(const _Float16* W, int kstride, int n0,
                                      int k0, int lane) {
  int n = n0 + (lane & 15);
  int k = k0 + ((lane >> 4) << 3);
  v8h lo = *(const v8h*)(W + n * kstride + k);
  v8h hi = *(const v8h*)(W + n * kstride + k + 16);
  v16h b;
#pragma unroll
  for (int i = 0; i < 8; ++i) { b[i] = lo[i]; b[i + 8] = hi[i]; }
  return b;
}

__device__ __forceinline__ v8f zero8() {
  v8f z;
#pragma unroll
  for (int i = 0; i < 8; ++i) z[i] = 0.f;
  return z;
}

#if USE_TDM
// ---------------------------------------------------------------------------
// Tensor Data Mover: 1-D tile load (nelem f16, one row) global -> LDS.
// D# per CDNA5 ISA: group0 = {count|flags, lds_addr,
// global_addr[31:0], global_addr[56:32]|type=2<<30}; group1 packs
// data_size=1 (2B), tensor_dim0=nelem, tensor_dim1=1, tile_dim0=nelem,
// tile_dim1=1, tensor_dim0_stride=nelem.
// ---------------------------------------------------------------------------
__device__ __forceinline__ void tdm_load_f16_row(unsigned lds_off,
                                                 const void* gsrc,
                                                 unsigned nelem) {
  unsigned long long ga = (unsigned long long)(uintptr_t)gsrc;
  u32x4 g0;
  g0[0] = 1u;                                     // count=1, user descriptor
  g0[1] = lds_off;                                // lds_addr (bytes)
  g0[2] = (unsigned)(ga & 0xffffffffu);           // global_addr[31:0]
  g0[3] = (unsigned)((ga >> 32) & 0x01ffffffu) | 0x80000000u;  // [56:32]|type=2
  i32x8 g1;
  g1[0] = (int)(1u << 16);                        // data_size=1 (2B), mask=0
  g1[1] = (int)((nelem & 0xffffu) << 16);         // tensor_dim0[15:0]
  g1[2] = (int)(((nelem >> 16) & 0xffffu) | (1u << 16));  // td0[31:16], td1=1
  g1[3] = (int)((nelem & 0xffffu) << 16);         // tile_dim0 = nelem
  g1[4] = 1;                                      // tile_dim1=1, tile_dim2=0
  g1[5] = (int)nelem;                             // tensor_dim0_stride lo32
  g1[6] = 0;                                      // stride hi, td1_stride lo
  g1[7] = 0;
  i32x4 gz;
  gz[0] = gz[1] = gz[2] = gz[3] = 0;
#if defined(__clang_major__) && __clang_major__ >= 23
  i32x8 gz8;
#pragma unroll
  for (int i = 0; i < 8; ++i) gz8[i] = 0;
  __builtin_amdgcn_tensor_load_to_lds(g0, g1, gz, gz, gz8, 0);
#else
  __builtin_amdgcn_tensor_load_to_lds(g0, g1, gz, gz, 0);
#endif
}
#endif  // USE_TDM

// ---------------------------------------------------------------------------
// Prep: convert residual-layer weights to f16, layer-contiguous [W1|W2] so a
// single TDM descriptor stages a whole layer (16 KB) into LDS.
// ---------------------------------------------------------------------------
__global__ void prep_weights_kernel(const float* __restrict__ Wres1,
                                    const float* __restrict__ Wres2,
                                    _Float16* __restrict__ wf16) {
  int i = blockIdx.x * blockDim.x + threadIdx.x;
  if (i >= NLAYER * 4096) return;
  int l = i >> 12, j = i & 4095;
  wf16[l * 8192 + j]        = (_Float16)Wres1[i];
  wf16[l * 8192 + 4096 + j] = (_Float16)Wres2[i];
}

// ---------------------------------------------------------------------------
// Fused MLP (WMMA) + rotation/projection.  8 waves/block, 16 rows per wave,
// 128 rows/block, 2500 blocks = 320000 rows.  Layer weights staged by TDM.
// ---------------------------------------------------------------------------
__global__ __launch_bounds__(256) void mlp_wmma_kernel(
    const float* __restrict__ z, const float* __restrict__ r,
    const float* __restrict__ pos, const float* __restrict__ W0,
    const float* __restrict__ Wres1, const float* __restrict__ Wres2,
    const float* __restrict__ W1a, const float* __restrict__ W1b,
    const _Float16* __restrict__ wres16, float* __restrict__ o_ppos,
    float* __restrict__ o_posd, float* __restrict__ o_res) {
  __shared__ _Float16 s_w0[64 * 32];              // W0 padded K 19->32, [N][K]
  __shared__ _Float16 s_w1a[16 * 64];             // W1a padded N 3->16, [N][K]
  __shared__ __align__(128) _Float16 s_wl[2 * 4096];  // TDM dest: [W1|W2]
  __shared__ _Float16 s_h[8][16 * 64];            // per-wave activations
  __shared__ _Float16 s_t[8][16 * 64];            // per-wave relu/tanh temp

  const int  tid     = threadIdx.x;
  const int  wave    = tid >> 5;
  const int  lane    = tid & 31;
  const long rowbase = (long)blockIdx.x * 128 + (long)wave * 16;

  // Stage constant weights once.
  for (int i = tid; i < 64 * 32; i += 256) {
    int n = i >> 5, k = i & 31;
    s_w0[i] = (_Float16)((k < 19) ? W0[n * 19 + k] : 0.f);
  }
  for (int i = tid; i < 16 * 64; i += 256) {
    int n = i >> 6, k = i & 63;
    s_w1a[i] = (_Float16)((n < 3) ? W1a[n * 64 + k] : 0.f);
  }

  // Build input tile X = [pos | z] padded to K=32 into s_h (cols 0..31).
  {
    int  row  = lane & 15;
    int  kb   = (lane >> 4) * 16;
    long grow = rowbase + row;
    int  b    = (int)(grow / NPTS);
    int  n    = (int)(grow % NPTS);
#pragma unroll
    for (int j = 0; j < 16; ++j) {
      int   k = kb + j;
      float v = 0.f;
      if (k < 3)       v = pos[n * 3 + k];
      else if (k < 19) v = z[b * LAT + (k - 3)];
      s_h[wave][row * 64 + k] = (_Float16)v;
    }
  }
  __syncthreads();

  const int col = lane & 15;
  const int rb8 = (lane >> 4) * 8;  // D-matrix row base for this lane half

  // ---- GEMM0: h = X @ W0^T  (K=32, one WMMA per N-tile) ----
  {
    v8f  acc[4];
    v16h a = afrag(&s_h[wave][0], 0, lane);
#pragma unroll
    for (int nn = 0; nn < 4; ++nn) {
      v16h b  = bfrag(s_w0, 32, nn * 16, 0, lane);
      acc[nn] = __builtin_amdgcn_wmma_f32_16x16x32_f16(
          false, a, false, b, (short)0, zero8(), false, false);
    }
#pragma unroll
    for (int nn = 0; nn < 4; ++nn)
#pragma unroll
      for (int g = 0; g < 8; ++g)
        s_h[wave][(rb8 + g) * 64 + nn * 16 + col] = (_Float16)acc[nn][g];
  }
  __syncthreads();

  // ---- 4 residual layers: h += relu(h @ W1^T) @ W2^T ----
  for (int l = 0; l < NLAYER; ++l) {
#if USE_TDM
    // One TDM descriptor stages the whole 16KB layer [W1|W2] into LDS;
    // wave 0 issues + waits on TENSORcnt, barrier publishes to the block.
    if (wave == 0) {
      tdm_load_f16_row((unsigned)(uintptr_t)(void*)&s_wl[0],
                       wres16 + (size_t)l * 8192, 8192u);
      __builtin_amdgcn_s_wait_tensorcnt(0);
    }
    __syncthreads();
#else
    for (int i = tid; i < 64 * 64; i += 256) {
      s_wl[i]        = (_Float16)Wres1[l * 4096 + i];
      s_wl[4096 + i] = (_Float16)Wres2[l * 4096 + i];
    }
    __syncthreads();
#endif
    const _Float16* s_w1 = s_wl;
    const _Float16* s_w2 = s_wl + 4096;

    // t = relu(h @ W1^T)
    {
      v8f acc[4];
#pragma unroll
      for (int nn = 0; nn < 4; ++nn) acc[nn] = zero8();
#pragma unroll
      for (int kk = 0; kk < 2; ++kk) {
        v16h a = afrag(&s_h[wave][0], kk * 32, lane);
#pragma unroll
        for (int nn = 0; nn < 4; ++nn) {
          v16h b  = bfrag(s_w1, 64, nn * 16, kk * 32, lane);
          acc[nn] = __builtin_amdgcn_wmma_f32_16x16x32_f16(
              false, a, false, b, (short)0, acc[nn], false, false);
        }
      }
#pragma unroll
      for (int nn = 0; nn < 4; ++nn)
#pragma unroll
        for (int g = 0; g < 8; ++g)
          s_t[wave][(rb8 + g) * 64 + nn * 16 + col] =
              (_Float16)fmaxf(acc[nn][g], 0.f);
    }
    __syncthreads();

    // h += t @ W2^T
    {
      v8f acc[4];
#pragma unroll
      for (int nn = 0; nn < 4; ++nn) acc[nn] = zero8();
#pragma unroll
      for (int kk = 0; kk < 2; ++kk) {
        v16h a = afrag(&s_t[wave][0], kk * 32, lane);
#pragma unroll
        for (int nn = 0; nn < 4; ++nn) {
          v16h b  = bfrag(s_w2, 64, nn * 16, kk * 32, lane);
          acc[nn] = __builtin_amdgcn_wmma_f32_16x16x32_f16(
              false, a, false, b, (short)0, acc[nn], false, false);
        }
      }
#pragma unroll
      for (int nn = 0; nn < 4; ++nn)
#pragma unroll
        for (int g = 0; g < 8; ++g) {
          int   idx = (rb8 + g) * 64 + nn * 16 + col;
          float cur = (float)s_h[wave][idx];
          s_h[wave][idx] = (_Float16)(cur + acc[nn][g]);
        }
    }
    __syncthreads();
  }

  // ---- head: t3 = tanh(h @ W1a^T)  (N padded 3->16) ----
  {
    v8f acc = zero8();
#pragma unroll
    for (int kk = 0; kk < 2; ++kk) {
      v16h a = afrag(&s_h[wave][0], kk * 32, lane);
      v16h b = bfrag(s_w1a, 64, 0, kk * 32, lane);
      acc    = __builtin_amdgcn_wmma_f32_16x16x32_f16(
          false, a, false, b, (short)0, acc, false, false);
    }
#pragma unroll
    for (int g = 0; g < 8; ++g)
      s_t[wave][(rb8 + g) * 64 + col] = (_Float16)tanhf(acc[g]);
  }
  __syncthreads();

  // ---- scalar tail: res = t3 @ W1b^T ; pos_d ; rotate ; project ----
  if (lane < 16) {
    int  row  = lane;
    long grow = rowbase + row;
    int  b    = (int)(grow / NPTS);
    int  n    = (int)(grow % NPTS);
    float t0 = (float)s_t[wave][row * 64 + 0];
    float t1 = (float)s_t[wave][row * 64 + 1];
    float t2 = (float)s_t[wave][row * 64 + 2];
    float res[3], pd[3];
#pragma unroll
    for (int j = 0; j < 3; ++j) {
      res[j] = t0 * W1b[j * 3 + 0] + t1 * W1b[j * 3 + 1] + t2 * W1b[j * 3 + 2];
      pd[j]  = pos[n * 3 + j] + res[j];
    }
    float ang_a = r[b * 3 + 0], ang_b = r[b * 3 + 1], ang_c = r[b * 3 + 2];
    float ca, sa, cb, sb, cc, sc;
    sincosf(ang_a, &sa, &ca);
    sincosf(ang_b, &sb, &cb);
    sincosf(ang_c, &sc, &cc);
    float R00 = ca * cb * cc - sa * sc, R01 = -ca * cb * sc - sa * cc;
    float R10 = sa * cb * cc + ca * sc, R11 = -sa * cb * sc + ca * cc;
    float R20 = -sb * cc,               R21 = sb * sc;
    float p0 = pd[0] * R00 + pd[1] * R10 + pd[2] * R20;
    float p1 = pd[0] * R01 + pd[1] * R11 + pd[2] * R21;
    o_ppos[grow * 2 + 0] = p0;
    o_ppos[grow * 2 + 1] = p1;
#pragma unroll
    for (int j = 0; j < 3; ++j) {
      o_posd[grow * 3 + j] = pd[j];
      o_res[grow * 3 + j]  = res[j];
    }
  }
}

// ---------------------------------------------------------------------------
// Bilinear scatter-splat with native float atomics (L2-resident images).
// ---------------------------------------------------------------------------
__global__ void splat_kernel(const float* __restrict__ ppos,
                             const float* __restrict__ amp,
                             const float* __restrict__ ampvar,
                             float* __restrict__ im) {
  long i = (long)blockIdx.x * blockDim.x + threadIdx.x;
  if (i >= (long)BSZ * NPTS) return;
  int b = (int)(i / NPTS);
  int n = (int)(i % NPTS);

  float a0 = ampvar[n], a1 = ampvar[NPTS + n];
  float mx = fmaxf(a0, a1);
  float e0 = __expf(a0 - mx), e1 = __expf(a1 - mx);
  float inv = amp[0] / (e0 + e1);
  float v0 = e0 * inv, v1 = e1 * inv;

  float c0 = ppos[i * 2 + 0], c1 = ppos[i * 2 + 1];
  float p0 = (c0 + 0.5f) * (float)(BOX - 1);
  float p1 = (c1 + 0.5f) * (float)(BOX - 1);
  float fl0 = floorf(p0), fl1 = floorf(p1);
  float f0 = p0 - fl0, f1 = p1 - fl1;
  int   i0 = (int)fl0, i1 = (int)fl1;

  float* im0 = im + ((long)b * 2 + 0) * (BOX * BOX);
  float* im1 = im + ((long)b * 2 + 1) * (BOX * BOX);
#pragma unroll
  for (int dy = 0; dy < 2; ++dy)
#pragma unroll
    for (int dx = 0; dx < 2; ++dx) {
      float w  = (dy ? f0 : 1.f - f0) * (dx ? f1 : 1.f - f1);
      int   yi = min(max(i0 + dy, 0), BOX - 1);
      int   xi = min(max(i1 + dx, 0), BOX - 1);
      long  o  = (long)yi * BOX + xi;
      unsafeAtomicAdd(im0 + o, v0 * w);
      unsafeAtomicAdd(im1 + o, v1 * w);
    }
}

// ---------------------------------------------------------------------------
// 256-point radix-2 DIT FFT in LDS (forward, exp(-2*pi*i*jk/N), unnormalized).
// ---------------------------------------------------------------------------
__device__ __forceinline__ void fft256(float2* buf, int tid) {
#pragma unroll
  for (int s = 1; s <= 8; ++s) {
    int half = 1 << (s - 1);
    int j    = tid & (half - 1);
    int base = (tid >> (s - 1)) << s;
    float sn, cs;
    sincospif(-(float)j / (float)half, &sn, &cs);
    float2 u = buf[base + j];
    float2 t = buf[base + j + half];
    float  tr = t.x * cs - t.y * sn;
    float  ti = t.x * sn + t.y * cs;
    buf[base + j]        = make_float2(u.x + tr, u.y + ti);
    buf[base + j + half] = make_float2(u.x - tr, u.y - ti);
    __syncthreads();
  }
}

__global__ __launch_bounds__(128) void fft_row_kernel(
    const float* __restrict__ im, float2* __restrict__ F) {
  __shared__ float2 buf[256];
  int  tid = threadIdx.x;
  long row = blockIdx.x;  // (b*2+c)*256 + y
  const float* src = im + row * 256;
  for (int e = tid; e < 256; e += 128) {
    int rev  = (int)(__brev((unsigned)e) >> 24);
    buf[rev] = make_float2(src[e], 0.f);
  }
  __syncthreads();
  fft256(buf, tid);
  float2* dst = F + row * 256;
  for (int e = tid; e < 256; e += 128) dst[e] = buf[e];
}

__global__ __launch_bounds__(128) void fft_col_kernel(float2* __restrict__ F) {
  __shared__ float2 buf[256];
  int tid = threadIdx.x;
  int bc  = blockIdx.x >> 8;   // image index 0..31
  int x   = blockIdx.x & 255;  // column
  float2* base = F + (long)bc * 65536 + x;
  for (int e = tid; e < 256; e += 128) {
    int rev  = (int)(__brev((unsigned)e) >> 24);
    buf[rev] = base[(long)e * 256];
  }
  __syncthreads();
  fft256(buf, tid);
  for (int e = tid; e < 256; e += 128) base[(long)e * 256] = buf[e];
}

// ---------------------------------------------------------------------------
// Apply analytic ifftshifted CTF filter and sum channels -> complex Proj.
// ---------------------------------------------------------------------------
__global__ void combine_kernel(const float2* __restrict__ F,
                               const float* __restrict__ A,
                               const float* __restrict__ Bc,
                               float* __restrict__ proj) {
  long i = (long)blockIdx.x * blockDim.x + threadIdx.x;
  if (i >= (long)BSZ * BOX * BOX) return;
  int b  = (int)(i >> 16);
  int ky = (int)((i >> 8) & 255);
  int kx = (int)(i & 255);
  float fy  = (ky < 128) ? (float)ky : (float)(ky - 256);
  float fx  = (kx < 128) ? (float)kx : (float)(kx - 256);
  float rad = rintf(sqrtf(fy * fy + fx * fx));
  float re = 0.f, imv = 0.f;
#pragma unroll
  for (int c = 0; c < 2; ++c) {
    float  ff = __expf(-Bc[c] * Bc[c] * rad) * A[c] * A[c];
    float2 v  = F[(((long)b * 2 + c) << 16) + ((long)ky << 8) + kx];
    re  += ff * v.x;
    imv += ff * v.y;
  }
  proj[i * 2 + 0] = re;
  proj[i * 2 + 1] = imv;
}

__global__ void zero_kernel(float* __restrict__ p, long n) {
  long i = (long)blockIdx.x * blockDim.x + threadIdx.x;
  for (; i < n; i += (long)gridDim.x * blockDim.x) p[i] = 0.f;
}

// ---------------------------------------------------------------------------
extern "C" void kernel_launch(void* const* d_in, const int* in_sizes, int n_in,
                              void* d_out, int out_size, void* d_ws,
                              size_t ws_size, hipStream_t stream) {
  const float* z      = (const float*)d_in[0];
  const float* r      = (const float*)d_in[1];
  const float* pos    = (const float*)d_in[2];
  const float* amp    = (const float*)d_in[3];
  const float* ampvar = (const float*)d_in[4];
  const float* W0     = (const float*)d_in[5];
  const float* Wres1  = (const float*)d_in[6];
  const float* Wres2  = (const float*)d_in[7];
  const float* W1a    = (const float*)d_in[8];
  const float* W1b    = (const float*)d_in[9];
  const float* A      = (const float*)d_in[10];
  const float* Bc     = (const float*)d_in[11];

  // Output layout (flat concat, complex64 interleaved re/im):
  // Proj(16*256*256 cplx) | proj_im(16*2*256*256) | proj_pos(16*20000*2)
  // | pos_d(16*20000*3) | res(16*20000*3)
  float* out    = (float*)d_out;
  float* o_proj = out;
  float* o_im   = out + 2097152;
  float* o_ppos = out + 4194304;
  float* o_posd = out + 4834304;
  float* o_res  = out + 5794304;

  // Workspace: FFT complex buffer (16.8 MB) then f16 layer weights (64 KB).
  float2*   F    = (float2*)d_ws;
  _Float16* wf16 = (_Float16*)((char*)d_ws + (size_t)16777216);

  zero_kernel<<<2048, 256, 0, stream>>>(out, (long)4194304);
  prep_weights_kernel<<<(NLAYER * 4096 + 255) / 256, 256, 0, stream>>>(
      Wres1, Wres2, wf16);
  mlp_wmma_kernel<<<2500, 256, 0, stream>>>(z, r, pos, W0, Wres1, Wres2, W1a,
                                            W1b, wf16, o_ppos, o_posd, o_res);
  splat_kernel<<<(BSZ * NPTS + 255) / 256, 256, 0, stream>>>(o_ppos, amp,
                                                             ampvar, o_im);
  fft_row_kernel<<<BSZ * NCH * BOX, 128, 0, stream>>>(o_im, F);
  fft_col_kernel<<<BSZ * NCH * BOX, 128, 0, stream>>>(F);
  combine_kernel<<<(BSZ * BOX * BOX + 255) / 256, 256, 0, stream>>>(F, A, Bc,
                                                                    o_proj);
}
